// EncoderBlock_35716948034028
// MI455X (gfx1250) — matrix operand order repaired and checked by reading, not compiled, not run on hardware
//
#include <hip/hip_runtime.h>
#include <hip/hip_bf16.h>

typedef __attribute__((ext_vector_type(16))) _Float16     v16h;
typedef __attribute__((ext_vector_type(8)))  float        v8f;
typedef __attribute__((ext_vector_type(4)))  unsigned int u32x4;
typedef __attribute__((ext_vector_type(8)))  int          i32x8;
typedef __attribute__((ext_vector_type(4)))  int          i32x4;
typedef unsigned int u32;

constexpr int B_ = 8, S_ = 1024, D_ = 1024, H_ = 16, DK_ = 64, DFF_ = 4096;

__device__ __forceinline__ int a_kbase(int v, int hi) {
    // 16-bit A-matrix 16x32 layout (ISA 7.12.2)
    return (v < 4 ? 2 * v : 16 + 2 * (v - 4)) + 8 * hi;
}
__device__ __forceinline__ int b_kbase(int v, int hi) {
    // 16-bit B-matrix 32x16 layout: same K order as A (ISA 7.12.2 table)
    return (v < 4 ? 2 * v : 16 + 2 * (v - 4)) + 8 * hi;
}
__device__ __forceinline__ v8f wmma_f16(v16h a, v16h b, v8f c) {
    return __builtin_amdgcn_wmma_f32_16x16x32_f16(
        /*neg_a=*/false, a, /*neg_b=*/false, b,
        /*c_mod=*/(short)0, c, /*reuse_a=*/false, /*reuse_b=*/false);
}

// ------------------------------------------------------------- TDM tile load
// 2D tile (tile_rows x tile_k f16 elements) from global (row stride
// stride_elems) into LDS at lds_addr, via Tensor Data Mover descriptor.
__device__ __forceinline__ void tdm_load_2d(unsigned lds_addr,
                                            const _Float16* gptr,
                                            unsigned tile_k, unsigned tile_rows,
                                            unsigned tensor_k,
                                            unsigned tensor_rows,
                                            unsigned stride_elems) {
    unsigned long long ga = (unsigned long long)(uintptr_t)gptr;
    u32x4 g0;
    g0[0] = 1u;                                   // count=1, user mode
    g0[1] = lds_addr;                             // lds_addr [63:32]
    g0[2] = (unsigned)(ga & 0xFFFFFFFFu);         // global_addr[31:0]
    g0[3] = (unsigned)((ga >> 32) & 0x01FFFFFFu)  // global_addr[56:32]
            | (2u << 30);                         // type=2 ("image")
    i32x8 g1;
    g1[0] = (int)(1u << 16);                      // data_size=1 (2 bytes)
    g1[1] = (int)((tensor_k & 0xFFFFu) << 16);    // tensor_dim0[15:0]
    g1[2] = (int)(((tensor_k >> 16) & 0xFFFFu) |  // tensor_dim0[31:16]
                  ((tensor_rows & 0xFFFFu) << 16));        // tensor_dim1 lo
    g1[3] = (int)(((tensor_rows >> 16) & 0xFFFFu) |        // tensor_dim1 hi
                  ((tile_k & 0xFFFFu) << 16));             // tile_dim0
    g1[4] = (int)(tile_rows & 0xFFFFu);                    // tile_dim1 (dim2=0)
    g1[5] = (int)stride_elems;                    // tensor_dim0_stride[31:0]
    g1[6] = 0;                                    // stride hi / dim1_stride lo
    g1[7] = 0;
    i32x4 z4;
    z4[0] = z4[1] = z4[2] = z4[3] = 0;
#if __clang_major__ >= 23
    i32x8 z8;
#pragma unroll
    for (int i = 0; i < 8; ++i) z8[i] = 0;
    __builtin_amdgcn_tensor_load_to_lds(g0, g1, z4, z4, z8, 0);
#else
    __builtin_amdgcn_tensor_load_to_lds(g0, g1, z4, z4, 0);
#endif
}

// ---------------------------------------------------------------- convert
__global__ void f32_to_f16_kernel(const float* __restrict__ in,
                                  _Float16* __restrict__ out, int n) {
    for (int i = blockIdx.x * blockDim.x + threadIdx.x; i < n;
         i += gridDim.x * blockDim.x)
        out[i] = (_Float16)in[i];
}

// in: [K][N] f32 row-major -> out: [N][K] f16 row-major (weight pre-transpose)
__global__ void transpose_f32_to_f16_kernel(const float* __restrict__ in,
                                            _Float16* __restrict__ out, int Kd,
                                            int Nd) {
    long long tot = (long long)Kd * Nd;
    for (long long idx = blockIdx.x * blockDim.x + threadIdx.x; idx < tot;
         idx += (long long)gridDim.x * blockDim.x) {
        int k = (int)(idx / Nd), n = (int)(idx % Nd);
        out[(size_t)n * Kd + k] = (_Float16)in[idx];
    }
}

// ---------------------------------------------------------------- GEMM
// C[M,N] = A[M,K] (f16 row-major) x Bt[N,K]^T (f16, pre-transposed) + bias
// BM=256 BN=64 BK=32; 256 threads = 8 waves; wave w owns rows [32w, 32w+32).
// Both tiles staged by the Tensor Data Mover, double-buffered.
#define GBM 256
#define GBN 64
#define GBK 32
__global__ __launch_bounds__(256) void gemm_wmma_kernel(
    const _Float16* __restrict__ A, const _Float16* __restrict__ Bt,
    const float* __restrict__ bias, _Float16* __restrict__ C16,
    float* __restrict__ C32, int M, int N, int K, int relu) {
    __shared__ _Float16 sA[2][GBM][GBK];  // 32 KB
    __shared__ _Float16 sB[2][GBN][GBK];  // 8 KB  (sB[n][k])
    const int tid = threadIdx.x, wave = tid >> 5, lane = tid & 31;
    const int m = lane & 15, hi = lane >> 4;
    const int bm = blockIdx.x * GBM, bn = blockIdx.y * GBN;

    v8f acc[2][4];
#pragma unroll
    for (int i = 0; i < 2; ++i)
#pragma unroll
        for (int j = 0; j < 4; ++j)
#pragma unroll
            for (int r = 0; r < 8; ++r) acc[i][j][r] = 0.f;

    auto issue = [&](int buf, int k0) {
        if (wave == 0) {
            tdm_load_2d((unsigned)(uintptr_t)&sA[buf][0][0],
                        A + (size_t)bm * K + k0, GBK, GBM, (unsigned)K,
                        (unsigned)M, (unsigned)K);
        } else if (wave == 1) {
            tdm_load_2d((unsigned)(uintptr_t)&sB[buf][0][0],
                        Bt + (size_t)bn * K + k0, GBK, GBN, (unsigned)K,
                        (unsigned)N, (unsigned)K);
        }
    };

    issue(0, 0);
    for (int k0 = 0; k0 < K; k0 += GBK) {
        const int cur = (k0 >> 5) & 1;
        if (k0 + GBK < K) {
            issue(cur ^ 1, k0 + GBK);               // prefetch next tile
            __builtin_amdgcn_s_wait_tensorcnt(1);   // oldest (cur) complete
        } else {
            __builtin_amdgcn_s_wait_tensorcnt(0);
        }
        __syncthreads();

        v16h afrag[2];
#pragma unroll
        for (int i = 0; i < 2; ++i) {
            u32* au = (u32*)&afrag[i];
#pragma unroll
            for (int v = 0; v < 8; ++v)
                au[v] = *(const u32*)(&sA[cur][wave * 32 + i * 16 + m]
                                         [a_kbase(v, hi)]);
        }
#pragma unroll
        for (int j = 0; j < 4; ++j) {
            v16h bfrag;
            u32* bu = (u32*)&bfrag;
#pragma unroll
            for (int v = 0; v < 8; ++v)
                bu[v] = *(const u32*)(&sB[cur][j * 16 + m][b_kbase(v, hi)]);
            acc[0][j] = wmma_f16(afrag[0], bfrag, acc[0][j]);
            acc[1][j] = wmma_f16(afrag[1], bfrag, acc[1][j]);
        }
        __syncthreads();
    }

    // epilogue: C layout row = r + 8*hi, col = lane&15
#pragma unroll
    for (int i = 0; i < 2; ++i)
#pragma unroll
        for (int j = 0; j < 4; ++j) {
            int col = bn + j * 16 + m;
            float bv = bias ? bias[col] : 0.f;
#pragma unroll
            for (int r = 0; r < 8; ++r) {
                int row = bm + wave * 32 + i * 16 + r + 8 * hi;
                float val = acc[i][j][r] + bv;
                if (relu) val = fmaxf(val, 0.f);
                if (C32) C32[(size_t)row * N + col] = val;
                if (C16) C16[(size_t)row * N + col] = (_Float16)val;
            }
        }
}

// ---------------------------------------------------------------- attention
// Flash attention. One wave per 16-query tile of one (b,h).
// Q,K,V,Out: [B,S,D] f16 with heads interleaved (D = H*DK).
__global__ __launch_bounds__(256) void attn_wmma_kernel(
    const _Float16* __restrict__ Q, const _Float16* __restrict__ Kh,
    const _Float16* __restrict__ Vh, const int* __restrict__ mask,
    _Float16* __restrict__ Out) {
    __shared__ _Float16 sP[8][16][32];  // wave-private P tiles, 8 KB
    const int tid = threadIdx.x, wave = tid >> 5, lane = tid & 31;
    const int m = lane & 15, hi = lane >> 4;
    const int gw = blockIdx.x * 8 + wave;  // 8192 waves total
    const int qt = gw & (S_ / 16 - 1);     // 64 q-tiles
    const int bh = gw >> 6;
    const int h = bh & (H_ - 1), b = bh >> 4;
    const int q0 = qt * 16;
    const size_t hoff = (size_t)h * DK_;

    // Q fragments: rows q0+m, dk chunks [0,32) and [32,64)
    v16h qf[2];
#pragma unroll
    for (int d2 = 0; d2 < 2; ++d2) {
        u32* qu = (u32*)&qf[d2];
#pragma unroll
        for (int v = 0; v < 8; ++v)
            qu[v] = *(const u32*)(Q + (size_t)(b * S_ + q0 + m) * D_ + hoff +
                                  d2 * 32 + a_kbase(v, hi));
    }

    v8f acc[4];
#pragma unroll
    for (int j = 0; j < 4; ++j)
#pragma unroll
        for (int r = 0; r < 8; ++r) acc[j][r] = 0.f;
    float mrow[8], lrow[8];
#pragma unroll
    for (int r = 0; r < 8; ++r) {
        mrow[r] = -3.0e38f;
        lrow[r] = 0.f;
    }

    for (int kt = 0; kt < S_; kt += 32) {
        if (kt + 32 < S_) {  // hint next K/V rows toward the caches
            __builtin_prefetch(Kh + (size_t)(b * S_ + kt + 32 + m) * D_ + hoff,
                               0, 0);
            __builtin_prefetch(Vh + (size_t)(b * S_ + kt + 32 + m) * D_ + hoff,
                               0, 0);
        }
        // ---- scores for keys [kt, kt+32): two 16x16 tiles
        v8f sc[2];
#pragma unroll
        for (int c2 = 0; c2 < 2; ++c2) {
            v8f s;
#pragma unroll
            for (int r = 0; r < 8; ++r) s[r] = 0.f;
            const int key = kt + c2 * 16 + m;  // this lane's key column
#pragma unroll
            for (int d2 = 0; d2 < 2; ++d2) {
                v16h kf;
                u32* ku = (u32*)&kf;
#pragma unroll
                for (int v = 0; v < 8; ++v)
                    ku[v] = *(const u32*)(Kh + (size_t)(b * S_ + key) * D_ +
                                          hoff + d2 * 32 + b_kbase(v, hi));
                s = wmma_f16(qf[d2], kf, s);
            }
            const int mv = mask[b * S_ + key];
#pragma unroll
            for (int r = 0; r < 8; ++r) {
                float t = s[r] * 0.125f;  // 1/sqrt(64)
                sc[c2][r] = (mv == 0) ? -1e9f : t;
            }
        }
        // ---- online softmax: row stats across the 16 lanes of each half
        float cmax[8], rsum[8];
#pragma unroll
        for (int r = 0; r < 8; ++r) cmax[r] = fmaxf(sc[0][r], sc[1][r]);
#pragma unroll
        for (int msk = 1; msk < 16; msk <<= 1)
#pragma unroll
            for (int r = 0; r < 8; ++r)
                cmax[r] = fmaxf(cmax[r], __shfl_xor(cmax[r], msk, 32));
        float mnew[8], scale[8];
#pragma unroll
        for (int r = 0; r < 8; ++r) {
            mnew[r] = fmaxf(mrow[r], cmax[r]);
            scale[r] = __expf(mrow[r] - mnew[r]);
            mrow[r] = mnew[r];
            rsum[r] = 0.f;
        }
        // P = exp(s - mnew): spill to LDS in A-fragment-friendly 16x32 tile
#pragma unroll
        for (int c2 = 0; c2 < 2; ++c2)
#pragma unroll
            for (int r = 0; r < 8; ++r) {
                float p = __expf(sc[c2][r] - mnew[r]);
                rsum[r] += p;
                sP[wave][r + 8 * hi][c2 * 16 + m] = (_Float16)p;
            }
#pragma unroll
        for (int msk = 1; msk < 16; msk <<= 1)
#pragma unroll
            for (int r = 0; r < 8; ++r)
                rsum[r] += __shfl_xor(rsum[r], msk, 32);
#pragma unroll
        for (int r = 0; r < 8; ++r) lrow[r] = lrow[r] * scale[r] + rsum[r];
#pragma unroll
        for (int j = 0; j < 4; ++j)
#pragma unroll
            for (int r = 0; r < 8; ++r) acc[j][r] *= scale[r];

        // ---- P (16x32 keys) x V (32 keys x 64 dk)
        v16h pf;
        {
            u32* pu = (u32*)&pf;
#pragma unroll
            for (int v = 0; v < 8; ++v)
                pu[v] = *(const u32*)(&sP[wave][m][a_kbase(v, hi)]);
        }
#pragma unroll
        for (int j = 0; j < 4; ++j) {
            v16h vf;
            _Float16* ve = (_Float16*)&vf;
#pragma unroll
            for (int v = 0; v < 8; ++v) {
                const int kb = b_kbase(v, hi);
                ve[2 * v] =
                    Vh[(size_t)(b * S_ + kt + kb) * D_ + hoff + j * 16 + m];
                ve[2 * v + 1] =
                    Vh[(size_t)(b * S_ + kt + kb + 1) * D_ + hoff + j * 16 + m];
            }
            acc[j] = wmma_f16(pf, vf, acc[j]);
        }
    }
    // ---- normalize and write out (f16, feeds O-projection GEMM)
#pragma unroll
    for (int j = 0; j < 4; ++j)
#pragma unroll
        for (int r = 0; r < 8; ++r) {
            float o = acc[j][r] / lrow[r];
            Out[(size_t)(b * S_ + q0 + r + 8 * hi) * D_ + hoff + j * 16 + m] =
                (_Float16)o;
        }
}

// ---------------------------------------------------------------- residual+LN
// out = alpha * (t - mean)/sqrt(var + eps) + beta,  t = x + y.  One WG per row.
__global__ __launch_bounds__(256) void resid_ln_kernel(
    const float* __restrict__ x, const float* __restrict__ y,
    const float* __restrict__ alpha, const float* __restrict__ beta,
    float* __restrict__ out32, _Float16* __restrict__ out16) {
    const int row = blockIdx.x, tid = threadIdx.x;
    const float* xr = x + (size_t)row * D_;
    const float* yr = y + (size_t)row * D_;
    __shared__ float red[8];
    float v[4];
    float s = 0.f;
#pragma unroll
    for (int j = 0; j < 4; ++j) {
        int i = tid + j * 256;
        float t = xr[i] + yr[i];
        v[j] = t;
        s += t;
    }
    for (int msk = 1; msk < 32; msk <<= 1) s += __shfl_xor(s, msk, 32);
    if ((tid & 31) == 0) red[tid >> 5] = s;
    __syncthreads();
    float tot = 0.f;
#pragma unroll
    for (int w = 0; w < 8; ++w) tot += red[w];
    const float mean = tot * (1.f / D_);
    __syncthreads();
    float s2 = 0.f;
#pragma unroll
    for (int j = 0; j < 4; ++j) {
        float d = v[j] - mean;
        s2 += d * d;
    }
    for (int msk = 1; msk < 32; msk <<= 1) s2 += __shfl_xor(s2, msk, 32);
    if ((tid & 31) == 0) red[tid >> 5] = s2;
    __syncthreads();
    float tot2 = 0.f;
#pragma unroll
    for (int w = 0; w < 8; ++w) tot2 += red[w];
    const float rstd = rsqrtf(tot2 * (1.f / D_) + 1e-6f);
#pragma unroll
    for (int j = 0; j < 4; ++j) {
        int i = tid + j * 256;
        float o = alpha[i] * (v[j] - mean) * rstd + beta[i];
        out32[(size_t)row * D_ + i] = o;
        if (out16) out16[(size_t)row * D_ + i] = (_Float16)o;
    }
}

// ---------------------------------------------------------------- launch
extern "C" void kernel_launch(void* const* d_in, const int* in_sizes, int n_in,
                              void* d_out, int out_size, void* d_ws,
                              size_t ws_size, hipStream_t stream) {
    const float* x    = (const float*)d_in[0];
    const int*   mask = (const int*)d_in[1];
    const float* wq = (const float*)d_in[2];
    const float* bq = (const float*)d_in[3];
    const float* wk = (const float*)d_in[4];
    const float* bk = (const float*)d_in[5];
    const float* wv = (const float*)d_in[6];
    const float* bv = (const float*)d_in[7];
    const float* wo = (const float*)d_in[8];
    const float* bo = (const float*)d_in[9];
    const float* ln1_a = (const float*)d_in[10];
    const float* ln1_b = (const float*)d_in[11];
    const float* ln2_a = (const float*)d_in[12];
    const float* ln2_b = (const float*)d_in[13];
    const float* w1 = (const float*)d_in[14];
    const float* b1 = (const float*)d_in[15];
    const float* w2 = (const float*)d_in[16];
    const float* b2 = (const float*)d_in[17];

    const int M = B_ * S_;  // 8192 rows
    char* ws = (char*)d_ws;
    size_t off = 0;
    auto take = [&](size_t bytes) -> char* {
        char* p = ws + off;
        off = (off + bytes + 255) & ~(size_t)255;
        return p;
    };
    _Float16* xh  = (_Float16*)take((size_t)M * D_ * 2);
    _Float16* wqt = (_Float16*)take((size_t)D_ * D_ * 2);    // [N][K]
    _Float16* wkt = (_Float16*)take((size_t)D_ * D_ * 2);
    _Float16* wvt = (_Float16*)take((size_t)D_ * D_ * 2);
    _Float16* wot = (_Float16*)take((size_t)D_ * D_ * 2);
    _Float16* w1t = (_Float16*)take((size_t)D_ * DFF_ * 2);  // [DFF][D]
    _Float16* w2t = (_Float16*)take((size_t)DFF_ * D_ * 2);  // [D][DFF]
    _Float16* Qh  = (_Float16*)take((size_t)M * D_ * 2);
    _Float16* Kh  = (_Float16*)take((size_t)M * D_ * 2);
    _Float16* Vh  = (_Float16*)take((size_t)M * D_ * 2);
    _Float16* ath = (_Float16*)take((size_t)M * D_ * 2);
    float*    prj = (float*)take((size_t)M * D_ * 4);
    float*    x1f = (float*)take((size_t)M * D_ * 4);
    _Float16* x1h = (_Float16*)take((size_t)M * D_ * 2);
    _Float16* ffh = (_Float16*)take((size_t)M * DFF_ * 2);
    float*    ff2 = (float*)take((size_t)M * D_ * 4);

    // 1) activations f32->f16; weights f32->f16 + transpose (for TDM staging)
    f32_to_f16_kernel<<<2048, 256, 0, stream>>>(x, xh, M * D_);
    transpose_f32_to_f16_kernel<<<1024, 256, 0, stream>>>(wq, wqt, D_, D_);
    transpose_f32_to_f16_kernel<<<1024, 256, 0, stream>>>(wk, wkt, D_, D_);
    transpose_f32_to_f16_kernel<<<1024, 256, 0, stream>>>(wv, wvt, D_, D_);
    transpose_f32_to_f16_kernel<<<1024, 256, 0, stream>>>(wo, wot, D_, D_);
    transpose_f32_to_f16_kernel<<<2048, 256, 0, stream>>>(w1, w1t, D_, DFF_);
    transpose_f32_to_f16_kernel<<<2048, 256, 0, stream>>>(w2, w2t, DFF_, D_);

    // 2) QKV projections (M=8192, N=1024, K=1024)
    dim3 gqkv(M / GBM, D_ / GBN);
    gemm_wmma_kernel<<<gqkv, 256, 0, stream>>>(xh, wqt, bq, Qh, nullptr, M, D_, D_, 0);
    gemm_wmma_kernel<<<gqkv, 256, 0, stream>>>(xh, wkt, bk, Kh, nullptr, M, D_, D_, 0);
    gemm_wmma_kernel<<<gqkv, 256, 0, stream>>>(xh, wvt, bv, Vh, nullptr, M, D_, D_, 0);

    // 3) flash attention: 8192 wave-tiles, 8 waves per block
    attn_wmma_kernel<<<(B_ * H_ * (S_ / 16)) / 8, 256, 0, stream>>>(Qh, Kh, Vh,
                                                                    mask, ath);

    // 4) output projection + residual LayerNorm 1
    gemm_wmma_kernel<<<gqkv, 256, 0, stream>>>(ath, wot, bo, nullptr, prj, M, D_, D_, 0);
    resid_ln_kernel<<<M, 256, 0, stream>>>(x, prj, ln1_a, ln1_b, x1f, x1h);

    // 5) FFN: relu(x1 @ w1 + b1) @ w2 + b2, then residual LayerNorm 2
    dim3 gff1(M / GBM, DFF_ / GBN);
    gemm_wmma_kernel<<<gff1, 256, 0, stream>>>(x1h, w1t, b1, ffh, nullptr, M, DFF_, D_, 1);
    gemm_wmma_kernel<<<gqkv, 256, 0, stream>>>(ffh, w2t, b2, nullptr, ff2, M, D_, DFF_, 0);
    resid_ln_kernel<<<M, 256, 0, stream>>>(x1f, ff2, ln2_a, ln2_b,
                                           (float*)d_out, nullptr);
}
